// ConvLRUBlock_55705725829900
// MI455X (gfx1250) — matrix operand order repaired and checked by reading, not compiled, not run on hardware
//
#include <hip/hip_runtime.h>
#include <math.h>

// ---------------------------------------------------------------------------
// ConvLRU block for MI455X (gfx1250, wave32, WMMA).
//
// Algebra: pad/fft96/proj/ifft/crop/fft64 collapses to h = FFT2_64(proj_c x);
// DFT matrices are folded into the rank projectors:
//   Ueff[c] = Fs^T conj(U[c])   (64x32)   Veff[c] = Fw^T V[c]      (64x32)
//   Urec[c] = IFs U[c]          (64x32)   Vrec[c] = IFw conj(V[c]) (64x32)
// Forward:  zq = Ueff^T (proj_c x) Veff        (per b,l,c)
// Scan:     z_l = lamb_l z_{l-1} + gamma_l zq_l
// Reverse:  h_sp = Urec z Vrec^T               (per b,l,c)
// Then conv3x3 (implicit-GEMM WMMA), proj3 (real GEMM), LayerNorm + residual.
// proj_b lands on a cropped-out delta row -> exactly zero.
// proj3_b is removed by LayerNorm mean-subtraction -> dropped.
// ---------------------------------------------------------------------------

typedef __attribute__((ext_vector_type(16))) _Float16 v16h;
typedef __attribute__((ext_vector_type(8)))  float    v8f;

#define PI_F 3.14159265358979323846f

#define Bx 4
#define Lx 32
#define Cx 32
#define Sx 64
#define Wx 64
#define Rx 32

// ---------------------------------------------------------------------------
// Batched complex GEMM on v_wmma_f32_16x16x32_f16, template-specialized on
// operand complexity. REQUIRES M,N,K multiples of 32 (true for all stages).
//
// LDS layout is fragment-major so each lane's WMMA fragment is 32 contiguous
// bytes (one v16h load -> two ds_load_b128, no VALU repacking):
//   A: row m, column slot k (lane reads k {0..15} or {16..31} directly)
//   B: stored transposed Bt[n][k] (lane reads k {0..15} or {16..31} directly)
// Row stride 48 halves (96B): 32B-aligned fragments, bank-spread.
// ---------------------------------------------------------------------------
struct CGemmArgs {
  const float *Are, *Aim, *Bre, *Bim;
  float *Dre, *Dim;
  int M, N, K;
  long sAm, sAk, sBk, sBn, sDm, sDn;
  long bA, bB, bD;     // batch strides (floats)
  int bAmod, bBmod;    // if >0: batch index is (p % mod) for that operand
};

template <int HA, int HB, int HD>
__global__ __launch_bounds__(128)
void cgemm_wmma(CGemmArgs g) {
  const int p = blockIdx.y;
  const int tilesN = g.N >> 5;
  const int tm = (blockIdx.x / tilesN) << 5;
  const int tn = (blockIdx.x % tilesN) << 5;
  const int pa = g.bAmod ? (p % g.bAmod) : p;
  const int pb = g.bBmod ? (p % g.bBmod) : p;

  const float* Are = g.Are + (long)pa * g.bA;
  const float* Aim = HA ? g.Aim + (long)pa * g.bA : nullptr;
  const float* Bre = g.Bre + (long)pb * g.bB;
  const float* Bim = HB ? g.Bim + (long)pb * g.bB : nullptr;
  float* Dre = g.Dre + (long)p * g.bD;
  float* Dim = HD ? g.Dim + (long)p * g.bD : nullptr;

  __shared__ _Float16 Asr[32][48];
  __shared__ _Float16 Asi[32][48];
  __shared__ _Float16 Bsr[32][48];
  __shared__ _Float16 Bsi[32][48];

  const int tid  = threadIdx.x;
  const int wave = tid >> 5;
  const int lane = tid & 31;
  const int sm   = (wave >> 1) << 4;   // subtile row base within 32x32 tile
  const int sn   = (wave & 1) << 4;    // subtile col base
  const int hs   = (lane >= 16) ? 1 : 0;
  const int fo   = hs << 4;            // fragment half-offset in LDS row
  const int am   = sm + (lane & 15);
  const int bn   = sn + (lane & 15);

  // Coalescing: walk the stride-1 axis with consecutive threads.
  const bool aMfast = (g.sAm == 1);
  const bool bNfast = (g.sBn == 1);

  v8f cre = {};
  v8f cim = {};

  for (int k0 = 0; k0 < g.K; k0 += 32) {
    for (int e = tid; e < 1024; e += 128) {
      // ---- A tile: element (m = tm+i, k = k0+j) -> Asr[i][j]
      {
        const int i = aMfast ? (e & 31) : (e >> 5);
        const int j = aMfast ? (e >> 5) : (e & 31);
        const long off = (long)(tm + i) * g.sAm + (long)(k0 + j) * g.sAk;
        const int slot = j;
        Asr[i][slot] = (_Float16)Are[off];
        if (HA) Asi[i][slot] = (_Float16)Aim[off];
      }
      // ---- B tile: element (k = k0+i, n = tn+j) -> Bt[j][i] (transposed)
      {
        const int i = bNfast ? (e >> 5) : (e & 31);
        const int j = bNfast ? (e & 31) : (e >> 5);
        const long off = (long)(k0 + i) * g.sBk + (long)(tn + j) * g.sBn;
        Bsr[j][i] = (_Float16)Bre[off];
        if (HB) Bsi[j][i] = (_Float16)Bim[off];
      }
    }
    __syncthreads();

    // Single vector load per fragment (two ds_load_b128 each).
    const v16h far = *(const v16h*)&Asr[am][fo];
    const v16h fbr = *(const v16h*)&Bsr[bn][fo];

    cre = __builtin_amdgcn_wmma_f32_16x16x32_f16(false, far, false, fbr, (short)0, cre, false, false);

    if (HA || HB) {
      v16h fai = {}, fbi = {};
      if (HA) fai = *(const v16h*)&Asi[am][fo];
      if (HB) fbi = *(const v16h*)&Bsi[bn][fo];
      if (HA && HB) {
        v16h fain;
        for (int e2 = 0; e2 < 16; ++e2) fain[e2] = -fai[e2];
        cre = __builtin_amdgcn_wmma_f32_16x16x32_f16(false, fain, false, fbi, (short)0, cre, false, false);
      }
      if (HD) {
        if (HB)
          cim = __builtin_amdgcn_wmma_f32_16x16x32_f16(false, far, false, fbi, (short)0, cim, false, false);
        if (HA)
          cim = __builtin_amdgcn_wmma_f32_16x16x32_f16(false, fai, false, fbr, (short)0, cim, false, false);
      }
    }
    __syncthreads();
  }

  // Scatter D (16x16 f32 C/D layout: VGPR v -> row v + 8*hs, col lane%16)
  for (int v = 0; v < 8; ++v) {
    const int m = tm + sm + v + (hs << 3);
    const int n = tn + sn + (lane & 15);
    const long off = (long)m * g.sDm + (long)n * g.sDn;
    Dre[off] = cre[v];
    if (HD) Dim[off] = cim[v];
  }
}

// ---------------------------------------------------------------------------
// 3x3 SAME conv as implicit GEMM on WMMA.
//   batch = (bl, plane): plane 0 = conv_r on Re(h_sp), plane 1 = conv_i on Im.
//   O[pix, co] = sum_{t=0..8, ci=0..31} in[ci][pix + shift(t)] * W[co][ci][t]
// K is processed as 9 chunks of 32: one tap over all 32 input channels, so
// the A tile is a shifted-window gather (border zeros) in fragment-major LDS.
// Output tile = 32 pixels x 32 co per block (4 waves, 2x2 of 16x16).
// ---------------------------------------------------------------------------
__global__ __launch_bounds__(128)
void conv_wmma(const float* __restrict__ hsp_re, const float* __restrict__ hsp_im,
               const float* __restrict__ wr, const float* __restrict__ brr,
               const float* __restrict__ wi, const float* __restrict__ bii,
               float* __restrict__ out) {
  const int pbat  = blockIdx.y;          // 0..255
  const int plane = pbat & 1;
  const long bl   = pbat >> 1;
  const float* in   = (plane ? hsp_im : hsp_re) + bl * 131072;
  const float* wgt  = plane ? wi : wr;   // (32co, 32ci, 3, 3)
  const float* bias = plane ? bii : brr;

  const int ptile = blockIdx.x << 5;     // 32 consecutive pixels (same row)
  const int py    = ptile >> 6;
  const int px0   = ptile & 63;

  __shared__ _Float16 Asr[32][48];
  __shared__ _Float16 Bsr[32][48];

  const int tid  = threadIdx.x;
  const int wave = tid >> 5;
  const int lane = tid & 31;
  const int sm   = (wave >> 1) << 4;
  const int sn   = (wave & 1) << 4;
  const int hs   = (lane >= 16) ? 1 : 0;
  const int fo   = hs << 4;
  const int am   = sm + (lane & 15);
  const int bn   = sn + (lane & 15);

  v8f acc = {};

  for (int t = 0; t < 9; ++t) {
    const int dy = t / 3 - 1;
    const int dx = t % 3 - 1;
    const int yy = py + dy;
    if (tid == 0 && yy >= 0 && yy < 64)
      __builtin_prefetch(in + yy * 64 + px0, 0, 1);   // global_prefetch_b8

    for (int e = tid; e < 1024; e += 128) {
      // A: pixel i (thread-fast, coalesced along x), channel j
      {
        const int i = e & 31;
        const int j = e >> 5;
        const int xx = px0 + i + dx;
        float v = 0.f;
        if (yy >= 0 && yy < 64 && xx >= 0 && xx < 64)
          v = in[j * 4096 + yy * 64 + xx];
        const int slot = j;
        Asr[i][slot] = (_Float16)v;
      }
      // B: k = ci, n = co -> Bt[co][ci]; weights are tiny (L2-resident)
      {
        const int ci = e >> 5;
        const int co = e & 31;
        Bsr[co][ci] = (_Float16)wgt[co * 288 + ci * 9 + t];
      }
    }
    __syncthreads();

    const v16h fa = *(const v16h*)&Asr[am][fo];
    const v16h fb = *(const v16h*)&Bsr[bn][fo];
    acc = __builtin_amdgcn_wmma_f32_16x16x32_f16(false, fa, false, fb, (short)0, acc, false, false);
    __syncthreads();
  }

  // Store: out[bl][plane*32+co][pix] = acc + bias[co]
  for (int v = 0; v < 8; ++v) {
    const int pix = ptile + sm + v + (hs << 3);
    const int co  = sn + (lane & 15);
    out[bl * 262144 + (long)(plane * 32 + co) * 4096 + pix] = acc[v] + bias[co];
  }
}

// ---------------------------------------------------------------------------
// Precompute folded projector matrices (per channel, 64x32 complex, planar).
// ---------------------------------------------------------------------------
__global__ void k_precompute(const float* __restrict__ Urow, const float* __restrict__ Vcol,
                             float* Ueff, float* Veff, float* Urec, float* Vrec) {
  const int idx = blockIdx.x * blockDim.x + threadIdx.x;   // C*64*32 = 65536
  if (idx >= Cx * 64 * 32) return;
  const int r = idx & 31;
  const int a = (idx >> 5) & 63;
  const int c = idx >> 11;
  float uer = 0.f, uei = 0.f, urr = 0.f, uri = 0.f;
  float ver = 0.f, vei = 0.f, vrr = 0.f, vri = 0.f;
  for (int s = 0; s < 64; ++s) {
    float sn, cs;
    __sincosf(2.f * PI_F * (float)(s * a) * (1.f / 64.f), &sn, &cs);
    const long ub = ((long)(c * 64 + s) * 32 + r) * 2;
    const float ur = Urow[ub], ui = Urow[ub + 1];
    const float vr = Vcol[ub], vi = Vcol[ub + 1];
    // f- = (cs, -sn) ; f+ = (cs, +sn)
    uer += cs * ur - sn * ui;   uei += -sn * ur - cs * ui;   // f- * conj(u)
    ver += cs * vr + sn * vi;   vei += -sn * vr + cs * vi;   // f- * v
    urr += cs * ur - sn * ui;   uri += sn * ur + cs * ui;    // f+ * u
    vrr += cs * vr + sn * vi;   vri += sn * vr - cs * vi;    // f+ * conj(v)
  }
  const float k = 0.125f;
  const long o = (long)c * 2048 + (long)a * 32 + r;
  const long P = (long)Cx * 2048;  // imag plane offset
  Ueff[o] = k * uer; Ueff[o + P] = k * uei;
  Veff[o] = k * ver; Veff[o + P] = k * vei;
  Urec[o] = k * urr; Urec[o + P] = k * uri;
  Vrec[o] = k * vrr; Vrec[o + P] = k * vri;
}

// ---------------------------------------------------------------------------
// ctx[b,l,c] = mean over (s,w) of x
// ---------------------------------------------------------------------------
__global__ void k_ctx(const float* __restrict__ x, float* ctx) {
  __shared__ float red[256];
  const long blc = blockIdx.x;
  const float* p = x + blc * 4096;
  float s = 0.f;
  for (int i = threadIdx.x; i < 4096; i += 256) s += p[i];
  red[threadIdx.x] = s;
  __syncthreads();
  for (int st = 128; st > 0; st >>= 1) {
    if ((int)threadIdx.x < st) red[threadIdx.x] += red[threadIdx.x + st];
    __syncthreads();
  }
  if (threadIdx.x == 0) ctx[blc] = red[0] * (1.f / 4096.f);
}

// ---------------------------------------------------------------------------
// Modulation MLP + lambda/gamma (per b,l)
// ---------------------------------------------------------------------------
__global__ void k_mod(const float* __restrict__ ctx, const float* __restrict__ listT,
                      const float* __restrict__ w1, const float* __restrict__ b1,
                      const float* __restrict__ w2, const float* __restrict__ b2,
                      const float* __restrict__ fscale, const float* __restrict__ plog,
                      const float* __restrict__ dmod,
                      float* lam_re, float* lam_im, float* gam) {
  const int bl = blockIdx.x;              // 128
  __shared__ float ctxs[33];
  __shared__ float hid[32];
  __shared__ float modv[2048];
  const int tid = threadIdx.x;            // 256
  if (tid < 32) ctxs[tid] = ctx[bl * 32 + tid];
  if (tid == 32) ctxs[32] = listT[bl];
  __syncthreads();
  if (tid < 32) {
    float a = b1[tid];
    for (int j = 0; j < 33; ++j) a += ctxs[j] * w1[tid * 33 + j];
    hid[tid] = tanhf(a);
  }
  __syncthreads();
  for (int o = tid; o < 2048; o += 256) {
    float a = b2[o];
    for (int j = 0; j < 32; ++j) a += hid[j] * w2[o * 32 + j];
    modv[o] = a;
  }
  __syncthreads();
  const float fs = fscale[0];
  const float dt = listT[bl];
  for (int cr = tid; cr < 1024; cr += 256) {
    const float dnu = fs * tanhf(modv[2 * cr]);
    const float dth = fs * tanhf(modv[2 * cr + 1]);
    const float nub = __expf(plog[cr] + dmod[cr]);
    const float thb = __expf(plog[1024 + cr] + dmod[1024 + cr]);
    const float nut = fmaxf(nub * dt + dnu, 1e-6f);
    const float tht = thb * dt + dth;
    const float e = __expf(-nut);
    float sn, cs;
    __sincosf(tht, &sn, &cs);
    const long o = (long)bl * 1024 + cr;
    lam_re[o] = e * cs;
    lam_im[o] = e * sn;
    gam[o] = sqrtf(fmaxf(1.f - __expf(-2.f * nut), 1e-12f));
  }
}

// ---------------------------------------------------------------------------
// LRU scan over L: z_l = lamb_l * z_{l-1} + gamma_l * zq_l   (z_{-1}=0)
// ---------------------------------------------------------------------------
__global__ void k_scan(const float* __restrict__ zq_re, const float* __restrict__ zq_im,
                       const float* __restrict__ lam_re, const float* __restrict__ lam_im,
                       const float* __restrict__ gam,
                       float* z_re, float* z_im, float* out_last) {
  const int idx = blockIdx.x * blockDim.x + threadIdx.x;   // B*C*R*Q = 131072
  if (idx >= Bx * Cx * Rx * Rx) return;
  const int q = idx & 31, r = (idx >> 5) & 31, c = (idx >> 10) & 31, b = idx >> 15;
  float zr = 0.f, zi = 0.f;
  for (int l = 0; l < Lx; ++l) {
    const long base = (((long)b * Lx + l) * Cx + c) * Rx + r;
    const long zoff = base * 32 + q;
    const float lr = lam_re[base], li = lam_im[base], g = gam[base];
    const float xr = zq_re[zoff] * g, xi = zq_im[zoff] * g;
    const float nzr = lr * zr - li * zi + xr;
    zi = lr * zi + li * zr + xi;
    zr = nzr;
    z_re[zoff] = zr;
    z_im[zoff] = zi;
  }
  const long lo = ((((long)b * Cx + c) * Rx + r) * 32 + q) * 2;
  out_last[lo] = zr;
  out_last[lo + 1] = zi;
}

// ---------------------------------------------------------------------------
// LayerNorm over (s,w) per (b,l,d), then residual add: out = x + LN(hf)
// ---------------------------------------------------------------------------
__global__ void k_ln(const float* __restrict__ hf, const float* __restrict__ x,
                     const float* __restrict__ lnw, const float* __restrict__ lnb,
                     float* out) {
  __shared__ float red[256], red2[256];
  const long bld = blockIdx.x;   // B*L*C = 4096
  const float* p = hf + bld * 4096;
  float s = 0.f, s2 = 0.f;
  for (int i = threadIdx.x; i < 4096; i += 256) {
    const float v = p[i];
    s += v;
    s2 += v * v;
  }
  red[threadIdx.x] = s;
  red2[threadIdx.x] = s2;
  __syncthreads();
  for (int st = 128; st > 0; st >>= 1) {
    if ((int)threadIdx.x < st) {
      red[threadIdx.x] += red[threadIdx.x + st];
      red2[threadIdx.x] += red2[threadIdx.x + st];
    }
    __syncthreads();
  }
  const float mu = red[0] * (1.f / 4096.f);
  const float var = red2[0] * (1.f / 4096.f) - mu * mu;
  const float inv = rsqrtf(var + 1e-5f);
  for (int i = threadIdx.x; i < 4096; i += 256) {
    const float v = (p[i] - mu) * inv * lnw[i] + lnb[i];
    out[bld * 4096 + i] = x[bld * 4096 + i] + v;
  }
}

// ---------------------------------------------------------------------------
// Host-side launch helper with template dispatch
// ---------------------------------------------------------------------------
static void launch_cgemm(hipStream_t st,
                         const float* Are, const float* Aim, long sAm, long sAk, long bA, int bAmod,
                         const float* Bre, const float* Bim, long sBk, long sBn, long bB, int bBmod,
                         float* Dre, float* Dim, long sDm, long sDn, long bD,
                         int M, int N, int K, int batch) {
  CGemmArgs g;
  g.Are = Are; g.Aim = Aim; g.Bre = Bre; g.Bim = Bim; g.Dre = Dre; g.Dim = Dim;
  g.M = M; g.N = N; g.K = K;
  g.sAm = sAm; g.sAk = sAk; g.sBk = sBk; g.sBn = sBn; g.sDm = sDm; g.sDn = sDn;
  g.bA = bA; g.bB = bB; g.bD = bD; g.bAmod = bAmod; g.bBmod = bBmod;
  dim3 grid((unsigned)((M / 32) * (N / 32)), (unsigned)batch);
  const bool ha = (Aim != nullptr), hb = (Bim != nullptr), hd = (Dim != nullptr);
  if (ha && hb && hd)       cgemm_wmma<1, 1, 1><<<grid, dim3(128), 0, st>>>(g);
  else if (!ha && hb && hd) cgemm_wmma<0, 1, 1><<<grid, dim3(128), 0, st>>>(g);
  else                      cgemm_wmma<0, 0, 0><<<grid, dim3(128), 0, st>>>(g);
}

extern "C" void kernel_launch(void* const* d_in, const int* in_sizes, int n_in,
                              void* d_out, int out_size, void* d_ws, size_t ws_size,
                              hipStream_t stream) {
  (void)in_sizes; (void)n_in; (void)out_size; (void)ws_size;

  const float* x      = (const float*)d_in[0];    // (B,L,C,S,W)
  const float* listT  = (const float*)d_in[1];    // (B,L)
  const float* plog   = (const float*)d_in[2];    // (2,C,R)
  const float* dmod   = (const float*)d_in[3];    // (2,C,R)
  const float* w1     = (const float*)d_in[4];    // (32,33)
  const float* b1     = (const float*)d_in[5];
  const float* w2     = (const float*)d_in[6];    // (2048,32)
  const float* b2     = (const float*)d_in[7];
  const float* fscale = (const float*)d_in[8];    // scalar
  const float* Urow   = (const float*)d_in[9];    // (C,S,R) complex interleaved
  const float* Vcol   = (const float*)d_in[10];   // (C,W,R) complex interleaved
  const float* projW  = (const float*)d_in[11];   // (C,C) complex interleaved
  /* d_in[12] proj_b: contributes only to a cropped-out delta row -> exactly 0 */
  const float* crw    = (const float*)d_in[13];
  const float* crb    = (const float*)d_in[14];
  const float* ciw    = (const float*)d_in[15];
  const float* cib    = (const float*)d_in[16];
  const float* p3w    = (const float*)d_in[17];   // (32,64)
  /* d_in[18] proj3_b: removed by LayerNorm mean subtraction */
  const float* lnw    = (const float*)d_in[19];
  const float* lnb    = (const float*)d_in[20];

  float* out = (float*)d_out;                     // 16777216 + 262144 floats

  // ---- workspace arena (floats), with lifetime-based overlays ----
  float* ws = (float*)d_ws;
  const long SZ_EFF = (long)Cx * 2048 * 2;        // 131072 each (re+im planar)
  float* Ueff   = ws;
  float* Veff   = Ueff + SZ_EFF;
  float* Urec   = Veff + SZ_EFF;
  float* Vrec   = Urec + SZ_EFF;
  float* ctx    = Vrec + SZ_EFF;                   // 4096
  float* lam_re = ctx + 4096;                      // 131072 each
  float* lam_im = lam_re + 131072;
  float* gam    = lam_im + 131072;
  // Region Y (33,554,432 floats): y -> hsp -> hfinal
  float* regY   = gam + 131072;
  float* y_re   = regY;
  float* y_im   = regY + 16777216;
  float* hsp_re = regY;                            // reuse after y dead
  float* hsp_im = regY + 16777216;
  float* hfin   = regY;                            // reuse after hsp dead
  // Region Z (33,554,432 floats): z1/tmp + zq + zs -> convout
  float* regZ   = regY + 33554432;
  float* z1_re  = regZ;                            // 8388608 each
  float* z1_im  = regZ + 8388608;
  float* zq_re  = regZ + 16777216;                 // 4194304 each
  float* zq_im  = regZ + 20971520;
  float* zs_re  = regZ + 25165824;
  float* zs_im  = regZ + 29360128;
  float* tmp_re = z1_re;                           // reuse after z1 dead
  float* tmp_im = z1_im;
  float* convo  = regZ;                            // reuse after recon done

  const long P = (long)Cx * 2048;                  // imag plane offset in *eff

  // 1) Folded projector matrices
  k_precompute<<<dim3(256), dim3(256), 0, stream>>>(Urow, Vcol, Ueff, Veff, Urec, Vrec);

  // 2) Channel projection: y[bl][d][sw] = sum_c x[bl][c][sw] * projW[c,d]
  launch_cgemm(stream,
               x, nullptr, /*sAm*/1, /*sAk*/4096, /*bA*/131072, 0,
               projW, projW + 1, /*sBk*/64, /*sBn*/2, /*bB*/0, 1,
               y_re, y_im, /*sDm*/1, /*sDn*/4096, /*bD*/131072,
               4096, 32, 32, Bx * Lx);

  // 3) Modulation path (independent of 2)
  k_ctx<<<dim3(Bx * Lx * Cx), dim3(256), 0, stream>>>(x, ctx);
  k_mod<<<dim3(Bx * Lx), dim3(256), 0, stream>>>(ctx, listT, w1, b1, w2, b2,
                                                 fscale, plog, dmod,
                                                 lam_re, lam_im, gam);

  // 4) z1[blc][r][w] = Ueff[c]^T(r,s) * y[blc](s,w)   M=32,N=64,K=64
  launch_cgemm(stream,
               Ueff, Ueff + P, /*sAm(r)*/1, /*sAk(s)*/32, /*bA*/2048, Cx,
               y_re, y_im, /*sBk(s)*/64, /*sBn(w)*/1, /*bB*/4096, 0,
               z1_re, z1_im, /*sDm*/64, /*sDn*/1, /*bD*/2048,
               32, 64, 64, Bx * Lx * Cx);

  // 5) zq[blc][r][q] = z1(r,w) * Veff[c](w,q)          M=32,N=32,K=64
  launch_cgemm(stream,
               z1_re, z1_im, 64, 1, 2048, 0,
               Veff, Veff + P, /*sBk(w)*/32, /*sBn(q)*/1, 2048, Cx,
               zq_re, zq_im, 32, 1, 1024,
               32, 32, 64, Bx * Lx * Cx);

  // 6) LRU scan over L (writes last_hidden tail of d_out)
  k_scan<<<dim3((Bx * Cx * Rx * Rx) / 256), dim3(256), 0, stream>>>(
      zq_re, zq_im, lam_re, lam_im, gam, zs_re, zs_im, out + 16777216);

  // 7) tmp[blc][r][w] = z(r,q) * Vrec[c]^T(q,w)        M=32,N=64,K=32
  launch_cgemm(stream,
               zs_re, zs_im, 32, 1, 1024, 0,
               Vrec, Vrec + P, /*sBk(q)*/1, /*sBn(w)*/32, 2048, Cx,
               tmp_re, tmp_im, 64, 1, 2048,
               32, 64, 32, Bx * Lx * Cx);

  // 8) h_sp[blc][s][w] = Urec[c](s,r) * tmp(r,w)       M=64,N=64,K=32
  launch_cgemm(stream,
               Urec, Urec + P, /*sAm(s)*/32, /*sAk(r)*/1, 2048, Cx,
               tmp_re, tmp_im, /*sBk(r)*/64, /*sBn(w)*/1, 2048, 0,
               hsp_re, hsp_im, 64, 1, 4096,
               64, 64, 32, Bx * Lx * Cx);

  // 9) 3x3 convs on Re/Im as implicit-GEMM WMMA -> 64-channel feature stack
  conv_wmma<<<dim3(128, 2 * Bx * Lx), dim3(128), 0, stream>>>(
      hsp_re, hsp_im, crw, crb, ciw, cib, convo);

  // 10) proj3: hf[bl][d][sw] = sum_k feat[bl][k][sw] * p3w[d,k]  (real GEMM)
  launch_cgemm(stream,
               convo, nullptr, /*sAm*/1, /*sAk*/4096, /*bA*/262144, 0,
               p3w, nullptr, /*sBk*/1, /*sBn*/64, /*bB*/0, 1,
               hfin, nullptr, /*sDm*/1, /*sDn*/4096, /*bD*/131072,
               4096, 32, 64, Bx * Lx);

  // 11) LayerNorm + residual
  k_ln<<<dim3(Bx * Lx * Cx), dim3(256), 0, stream>>>(hfin, x, lnw, lnb, out);
}
